// PreprocessPositions_40175124086930
// MI455X (gfx1250) — matrix operand order repaired and checked by reading, not compiled, not run on hardware
//
#include <hip/hip_runtime.h>
#include <hip/hip_bf16.h>
#include <math.h>

typedef __attribute__((ext_vector_type(16))) _Float16 v16h;
typedef __attribute__((ext_vector_type(8)))  _Float16 v8h;
typedef __attribute__((ext_vector_type(8)))  float    v8f;
typedef unsigned int u32x4 __attribute__((ext_vector_type(4)));
typedef int          i32x8 __attribute__((ext_vector_type(8)));
typedef int          i32x4 __attribute__((ext_vector_type(4)));

#define NK    32
#define NH    32
#define ED    768
#define BATCH 32
#define NN    256

// A-matrix 16x32 f16 layout (ISA 7.12.2): lane half h, VGPR pair v ->
// K = {2v, 2v+1} + 8h for v<4 ; {16+2(v-4), ...} + 8h for v>=4
__device__ __forceinline__ int abase(int v, int h) {
  return ((v < 4) ? (2 * v) : (8 + 2 * v)) + 8 * h;
}

__device__ __forceinline__ float gelu_exact(float x) {
  return 0.5f * x * (1.0f + erff(x * 0.70710678118654752440f));
}

// One workgroup per (b, i). 8 waves; each wave does 2 tiles of 16 j-values.
// pos row staged LDS via TDM; per tile: RBF -> WMMA(W1^T) -> GELU -> LDS
// reshuffle -> WMMA(W2^T) -> NT attn store.
__global__ __launch_bounds__(256) void gauss_bias_kernel(
    const float* __restrict__ pos,  const float* __restrict__ means,
    const float* __restrict__ stds, const float* __restrict__ W1,
    const float* __restrict__ b1,   const float* __restrict__ W2,
    const float* __restrict__ b2,   float* __restrict__ attn,
    float* __restrict__ fs)
{
  __shared__ __align__(16) float    posbuf[NN * 3];     // pos[b, :, :]
  __shared__ __align__(32) _Float16 hbuf[8][16 * 32];   // per-wave [pair][feat]
  __shared__ float fsum[NK];

  const int wg   = blockIdx.x;
  const int b    = wg >> 8;
  const int i    = wg & 255;
  const int tid  = threadIdx.x;
  const int wave = tid >> 5;
  const int lane = tid & 31;
  const int p    = lane & 15;   // pair / column index within tile
  const int hL   = lane >> 4;   // lane half

  // --- TDM: DMA pos[b,:,:] (768 f32, 1-D tile) into LDS, wave 0 issues ---
  if (wave == 0) {
    unsigned long long ga = (unsigned long long)(pos + (size_t)b * NN * 3);
    unsigned int lds_addr = (unsigned int)(size_t)(void*)posbuf;
    u32x4 g0;
    g0[0] = 1u;                                     // count=1 valid descriptor
    g0[1] = lds_addr;                               // lds_addr
    g0[2] = (unsigned int)ga;                       // global_addr[31:0]
    g0[3] = (unsigned int)((ga >> 32) & 0x1FFFFFFu) | (2u << 30); // addr hi | type=2
    i32x8 g1;
    g1[0] = (2 << 16);                              // data_size = 4 bytes
    g1[1] = (int)((NN * 3) << 16);                  // tensor_dim0[15:0]
    g1[2] = (int)(1u << 16);                        // dim0 hi=0 | tensor_dim1=1
    g1[3] = (int)((unsigned)(NN * 3) << 16);        // dim1 hi=0 | tile_dim0=768
    g1[4] = 0;                                      // tile_dim1=0, tile_dim2=0
    g1[5] = NN * 3;                                 // tensor_dim0_stride lo
    g1[6] = 0;
    g1[7] = 0;
    i32x4 gz4 = {0, 0, 0, 0};
    i32x8 gz8 = {0, 0, 0, 0, 0, 0, 0, 0};
    __builtin_amdgcn_tensor_load_to_lds(g0, g1, gz4, gz4, gz8, 0);
    __builtin_amdgcn_s_wait_tensorcnt(0);
  }
  if (tid < NK) fsum[tid] = 0.0f;

  // Per-lane RBF params for its 16 K values (B-fragment layout: K = 16*hL + v)
  float nmean_[16], istd_[16], coef_[16];
#pragma unroll
  for (int v = 0; v < 16; ++v) {
    int k = 16 * hL + v;
    float s = fabsf(stds[k]) + 0.01f;
    istd_[v]  = 1.0f / s;
    nmean_[v] = -means[k] / s;            // z = dist*istd + nmean (one FMA)
    coef_[v]  = 0.3989422804014327f / s;  // 1/(sqrt(2*pi)*std)
  }

  // Loop-invariant A fragments: rows of W1^T / W2^T (M = output channel)
  v16h a1lo, a1hi, a2lo, a2hi;
#pragma unroll
  for (int idx = 0; idx < 16; ++idx) {
    int K = 16 * hL + idx;
    a1lo[idx] = (_Float16)W1[K * NK + p];
    a1hi[idx] = (_Float16)W1[K * NK + p + 16];
    a2lo[idx] = (_Float16)W2[K * NH + p];
    a2hi[idx] = (_Float16)W2[K * NH + p + 16];
  }
  // Bias folded into WMMA C operand. D row M = r + 8*hL (+16 for hi tile).
  v8f cb1lo, cb1hi, cb2lo, cb2hi;
#pragma unroll
  for (int r = 0; r < 8; ++r) {
    int f = r + 8 * hL;
    cb1lo[r] = b1[f];  cb1hi[r] = b1[f + 16];
    cb2lo[r] = b2[f];  cb2hi[r] = b2[f + 16];
  }

  __syncthreads();   // covers TDM completion (wave0 waited) + fsum init

  const float xi = posbuf[i * 3 + 0];
  const float yi = posbuf[i * 3 + 1];
  const float zi = posbuf[i * 3 + 2];

  float facc[16];
#pragma unroll
  for (int v = 0; v < 16; ++v) facc[v] = 0.0f;

  _Float16* hb = &hbuf[wave][0];

  for (int t = wave; t < 16; t += 8) {
    const int j = t * 16 + p;
    float dx = xi - posbuf[j * 3 + 0];
    float dy = yi - posbuf[j * 3 + 1];
    float dz = zi - posbuf[j * 3 + 2];
    float sq = dx * dx + dy * dy + dz * dz;
    // v_sqrt_f32: sqrt(0)==0, so the reference's zero-guard is free here;
    // IEEE-exact sqrt is pointless ahead of the f16 truncation.
    float dist = __builtin_amdgcn_sqrtf(sq);

    // RBF values directly in B-fragment layout (col = pair p, K = 16*hL + v)
    v16h bq;
#pragma unroll
    for (int v = 0; v < 16; ++v) {
      float z = __builtin_fmaf(dist, istd_[v], nmean_[v]);
      float g = __expf(-0.5f * z * z) * coef_[v];   // v_exp_f32 path
      facc[v] += g;
      bq[v] = (_Float16)g;
    }

    // h = gelu(gk @ W1 + b1), computed transposed: D[f, pair]
    v8f d1lo = __builtin_amdgcn_wmma_f32_16x16x32_f16(false, a1lo, false, bq,
                                                      (short)0, cb1lo, false, false);
    v8f d1hi = __builtin_amdgcn_wmma_f32_16x16x32_f16(false, a1hi, false, bq,
                                                      (short)0, cb1hi, false, false);

    // GELU + stash to per-wave LDS tile [pair][feat]; each half is a
    // contiguous, 16B-aligned 8-half run -> single ds_store_b128 each.
    v8h lo8, hi8;
#pragma unroll
    for (int r = 0; r < 8; ++r) {
      lo8[r] = (_Float16)gelu_exact(d1lo[r]);
      hi8[r] = (_Float16)gelu_exact(d1hi[r]);
    }
    *(v8h*)(hb + p * 32 + 8 * hL)      = lo8;   // f = 8*hL + r
    *(v8h*)(hb + p * 32 + 16 + 8 * hL) = hi8;   // f = 16 + 8*hL + r
    asm volatile("s_wait_dscnt 0" ::: "memory"); // wave-internal LDS RAW

    v16h bh = *(const v16h*)(hb + p * 32 + 16 * hL); // col p, K = 16*hL + idx

    // attn_bias^T tile: D[ch, pair]
    v8f d2lo = __builtin_amdgcn_wmma_f32_16x16x32_f16(false, a2lo, false, bh,
                                                      (short)0, cb2lo, false, false);
    v8f d2hi = __builtin_amdgcn_wmma_f32_16x16x32_f16(false, a2hi, false, bh,
                                                      (short)0, cb2hi, false, false);

    // attn[b][ch0+.][i][j]: one per-lane base, immediate channel offsets,
    // non-temporal (write-once stream, don't pollute caches)
    float* outp = attn + (((size_t)b * NH + 8 * hL) * NN + i) * NN + j;
#pragma unroll
    for (int r = 0; r < 8; ++r) {
      __builtin_nontemporal_store(d2lo[r], outp + (size_t)r * (NN * NN));
      __builtin_nontemporal_store(d2hi[r], outp + (size_t)(16 + r) * (NN * NN));
    }
    asm volatile("s_wait_dscnt 0" ::: "memory"); // WAR guard before next tile
  }

  // feat_sum[b,i,k] = sum_j gk : LDS f32 atomics across the workgroup
#pragma unroll
  for (int v = 0; v < 16; ++v)
    atomicAdd(&fsum[16 * hL + v], facc[v]);
  __syncthreads();
  if (tid < NK) fs[(size_t)(b * NN + i) * NK + tid] = fsum[tid];
}

// node_feature = feat_sum[8192,32] @ node_W[32,768] + node_b.
// Each wave owns one 16-wide column strip (fixed nt): the node_W B-fragment
// and bias C are loaded once and reused across 16 row tiles.
#define MT_PER_WAVE 16
__global__ __launch_bounds__(256) void node_proj_kernel(
    const float* __restrict__ fs, const float* __restrict__ node_W,
    const float* __restrict__ node_b, float* __restrict__ outN)
{
  const int tid  = threadIdx.x;
  const int lane = tid & 31;
  const int p    = lane & 15;
  const int hL   = lane >> 4;
  const int w    = blockIdx.x * 8 + (tid >> 5);
  const int nt   = w % (ED / 16);
  const int mtg  = w / (ED / 16);
  const int e    = nt * 16 + p;

  v16h bm;                                 // loaded once per wave
#pragma unroll
  for (int idx = 0; idx < 16; ++idx) {
    int K = (idx < 8) ? (8 * hL + idx) : (16 + 8 * hL + (idx - 8));
    bm[idx] = (_Float16)node_W[(size_t)K * ED + e];
  }
  const float bias = node_b[e];            // per-column, same for all rows
  v8f c;
#pragma unroll
  for (int r = 0; r < 8; ++r) c[r] = bias;

  for (int it = 0; it < MT_PER_WAVE; ++it) {
    const int row0 = (mtg * MT_PER_WAVE + it) * 16;

    // Lane's K-set is two contiguous 8-float runs of its row: 4x b128 loads.
    const float4* rowp = (const float4*)(fs + (size_t)(row0 + p) * NK + 8 * hL);
    float tmpA[16];
    *(float4*)&tmpA[0]  = rowp[0];
    *(float4*)&tmpA[4]  = rowp[1];
    *(float4*)&tmpA[8]  = rowp[4];   // +16 floats
    *(float4*)&tmpA[12] = rowp[5];
    v16h a;
#pragma unroll
    for (int idx = 0; idx < 16; ++idx) a[idx] = (_Float16)tmpA[idx];

    v8f d = __builtin_amdgcn_wmma_f32_16x16x32_f16(false, a, false, bm,
                                                   (short)0, c, false, false);

    float* outp = outN + (size_t)(row0 + 8 * hL) * ED + e;
#pragma unroll
    for (int r = 0; r < 8; ++r)
      __builtin_nontemporal_store(d[r], outp + (size_t)r * ED);
  }
}

extern "C" void kernel_launch(void* const* d_in, const int* in_sizes, int n_in,
                              void* d_out, int out_size, void* d_ws, size_t ws_size,
                              hipStream_t stream) {
  const float* pos   = (const float*)d_in[0];
  const float* means = (const float*)d_in[1];
  const float* stds  = (const float*)d_in[2];
  const float* W1    = (const float*)d_in[3];
  const float* b1    = (const float*)d_in[4];
  const float* W2    = (const float*)d_in[5];
  const float* b2    = (const float*)d_in[6];
  const float* nW    = (const float*)d_in[7];
  const float* nb    = (const float*)d_in[8];

  float* attn = (float*)d_out;
  float* outN = (float*)d_out + (size_t)BATCH * NH * NN * NN;
  float* fs   = (float*)d_ws; // 8192*32 f32 = 1 MB scratch

  gauss_bias_kernel<<<BATCH * NN, 256, 0, stream>>>(pos, means, stds, W1, b1,
                                                    W2, b2, attn, fs);

  // waves = (8192/16/MT_PER_WAVE) * (768/16) = 32 * 48 = 1536 -> 192 blocks
  const int blocks = (BATCH * NN / 16 / MT_PER_WAVE) * (ED / 16) / 8;
  node_proj_kernel<<<blocks, 256, 0, stream>>>(fs, nW, nb, outN);

  (void)in_sizes; (void)n_in; (void)out_size; (void)ws_size;
}
